// HomophilicNodeClassifier_42339787604747
// MI455X (gfx1250) — matrix operand order repaired and checked by reading, not compiled, not run on hardware
//
#include <hip/hip_runtime.h>
#include <hip/hip_bf16.h>
#include <math.h>

// MI455X / gfx1250, wave32. bf16 WMMA GEMMs (f32 accum), L2-resident edge scatter.

typedef __attribute__((ext_vector_type(16))) __bf16 v16bf;
typedef __attribute__((ext_vector_type(8)))  float  v8f;

#define WAVE_SZ 32

// -------------------------------------------------------------------------
// Tiled GEMM: out[N x NPAD] = A[N x K](f32) @ W[K x NSRC](f32)
// W is packed into LDS in *fragment order* (bf16, zero-padded cols) so each
// lane's 16-element B operand is one contiguous 32B LDS read (2x ds_load_b128).
// A rows are converted f32->bf16 on the fly from two 32B global loads/lane.
// Per block: 32 rows, 8 waves; each wave owns one 16x16 output tile.
// MODE: 0 = plain store, 1 = +bias, 2 = +bias then ReLU.
// -------------------------------------------------------------------------
template<int K, int NPAD, int NSRC, int MODE>
__global__ __launch_bounds__(256) void gemm_wmma_bf16(
    const float* __restrict__ A, const float* __restrict__ W,
    const float* __restrict__ bias, float* __restrict__ out, int N)
{
    constexpr int NT    = NPAD / 16;   // col tiles
    constexpr int KS    = K / 32;      // k steps
    constexpr int TILES = 2 * NT;      // 32 rows per block

    __shared__ __bf16 Wpack[K * NPAD]; // max 16 KB (K=128, NPAD=64)

    const int tid  = threadIdx.x;
    const int lane = tid & (WAVE_SZ - 1);
    const int wave = tid >> 5;

    // Cooperative pack: Wpack[((ks*NT + ct)*32 + l)*16 + j] =
    //   bf16( W[ks*32 + (l>>4)*8 + (j&7) + (j>>3)*16][ct*16 + (l&15)] ), zero-pad cols >= NSRC.
    for (int idx = tid; idx < K * NPAD; idx += 256) {
        const int j    = idx & 15;
        const int l    = (idx >> 4) & 31;
        const int ctks = idx >> 9;
        const int ct   = ctks % NT;
        const int ks   = ctks / NT;
        const int row  = ks * 32 + ((l >> 4) << 3) + (j & 7) + ((j >> 3) << 4);
        const int col  = ct * 16 + (l & 15);
        float wv;
        if (NSRC < NPAD)
            wv = (col < NSRC) ? W[row * NSRC + col] : 0.0f;
        else
            wv = W[row * NSRC + col];
        Wpack[idx] = (__bf16)wv;
    }
    __syncthreads();

    if (TILES < 8 && wave >= TILES) return;

    const int rt = wave / NT;
    const int ct = wave - rt * NT;
    const int r0 = blockIdx.x * 32 + rt * 16;
    if (r0 >= N) return;
    const bool full = (r0 + 16 <= N);

    const int mrow = lane & 15;
    int rowA = r0 + mrow;
    if (!full && rowA >= N) rowA = N - 1;      // clamp for ragged tail
    const float* __restrict__ arow = A + (size_t)rowA * K;

    const int hi16 = lane >> 4;
    const int ncol = ct * 16 + (lane & 15);
    const __bf16* __restrict__ bfrag = Wpack + ((size_t)ct * 32 + lane) * 16;

    v8f acc = {0.f,0.f,0.f,0.f,0.f,0.f,0.f,0.f};

#pragma unroll
    for (int ks = 0; ks < KS; ++ks) {
        const int kb = ks * 32 + hi16 * 8;
        // A fragment (16-bit A layout: lane<16 -> K=k..k+7 & k+16..k+23;
        // lane>=16 -> K=k+8..k+15 & k+24..k+31), f32 -> bf16 on the fly.
        const float4 f0 = *(const float4*)(arow + kb);
        const float4 f1 = *(const float4*)(arow + kb + 4);
        const float4 f2 = *(const float4*)(arow + kb + 16);
        const float4 f3 = *(const float4*)(arow + kb + 20);
        v16bf a;
        a[0]=(__bf16)f0.x;  a[1]=(__bf16)f0.y;  a[2]=(__bf16)f0.z;  a[3]=(__bf16)f0.w;
        a[4]=(__bf16)f1.x;  a[5]=(__bf16)f1.y;  a[6]=(__bf16)f1.z;  a[7]=(__bf16)f1.w;
        a[8]=(__bf16)f2.x;  a[9]=(__bf16)f2.y;  a[10]=(__bf16)f2.z; a[11]=(__bf16)f2.w;
        a[12]=(__bf16)f3.x; a[13]=(__bf16)f3.y; a[14]=(__bf16)f3.z; a[15]=(__bf16)f3.w;

        // B fragment: one contiguous 32B LDS read.
        const v16bf b = *(const v16bf*)(bfrag + (size_t)ks * NT * 512);

        acc = __builtin_amdgcn_wmma_f32_16x16x32_bf16(
                  false, a, false, b, (short)0, acc, false, false);
    }

    float bv = 0.0f;
    if (MODE >= 1) {
        if (NSRC < NPAD) bv = (ncol < NSRC) ? bias[ncol] : 0.0f;
        else             bv = bias[ncol];
    }

    // C/D layout: VGPR i -> M = i (lanes 0-15) or M = 8+i (lanes 16-31).
    float* __restrict__ orow = out + (size_t)(r0 + hi16 * 8) * NPAD + ncol;
    if (full) {
#pragma unroll
        for (int i = 0; i < 8; ++i) {
            float v = acc[i] + bv;
            if (MODE == 2) v = fmaxf(v, 0.0f);
            orow[(size_t)i * NPAD] = v;
        }
    } else {
#pragma unroll
        for (int i = 0; i < 8; ++i) {
            const int row = r0 + hi16 * 8 + i;
            if (row < N) {
                float v = acc[i] + bv;
                if (MODE == 2) v = fmaxf(v, 0.0f);
                orow[(size_t)i * NPAD] = v;
            }
        }
    }
}

// -------------------------------------------------------------------------
// Degree / normalization kernels
// -------------------------------------------------------------------------
__global__ void deg_init_kernel(float* __restrict__ deg, int N)
{
    int i = blockIdx.x * blockDim.x + threadIdx.x;
    if (i < N) deg[i] = 1.0f;                  // self-loop contribution
}

__global__ void deg_accum_kernel(const int* __restrict__ dst,
                                 float* __restrict__ deg, int E)
{
    int e = blockIdx.x * blockDim.x + threadIdx.x;
    if (e < E) atomicAdd(&deg[dst[e]], 1.0f);
}

__global__ void dinv_kernel(const float* __restrict__ deg,
                            float* __restrict__ dinv, int N)
{
    int i = blockIdx.x * blockDim.x + threadIdx.x;
    if (i < N) dinv[i] = rsqrtf(deg[i]);       // deg >= 1 always
}

// agg[i][:] = h[i][:] * dinv[i]^2   (self-loop term; also initializes agg)
__global__ void selfloop_init_kernel(const float* __restrict__ h,
                                     const float* __restrict__ dinv,
                                     float* __restrict__ agg,
                                     int N, int chunks /* H/4 */)
{
    int t = blockIdx.x * blockDim.x + threadIdx.x;
    if (t >= N * chunks) return;
    const int i = t / chunks;
    const int c = (t - i * chunks) * 4;
    const float w = dinv[i] * dinv[i];
    const float4 hv = *(const float4*)(h + (size_t)i * (chunks * 4) + c);
    float4 o; o.x = hv.x * w; o.y = hv.y * w; o.z = hv.z * w; o.w = hv.w * w;
    *(float4*)(agg + (size_t)i * (chunks * 4) + c) = o;
}

// agg[dst][:] += h[src][:] * dinv[src]*dinv[dst]  (L2-resident atomics)
__global__ void edge_scatter_kernel(const int* __restrict__ src,
                                    const int* __restrict__ dst,
                                    const float* __restrict__ dinv,
                                    const float* __restrict__ h,
                                    float* __restrict__ agg,
                                    int E, int chunks /* H/4 */)
{
    int t = blockIdx.x * blockDim.x + threadIdx.x;
    if (t >= E * chunks) return;
    const int e = t / chunks;
    const int c = (t - e * chunks) * 4;
    const int s = src[e];
    const int d = dst[e];
    const float w = dinv[s] * dinv[d];
    const int H = chunks * 4;
    const float4 hv = *(const float4*)(h + (size_t)s * H + c);
    float* o = agg + (size_t)d * H + c;
    atomicAdd(o + 0, hv.x * w);
    atomicAdd(o + 1, hv.y * w);
    atomicAdd(o + 2, hv.z * w);
    atomicAdd(o + 3, hv.w * w);
}

// g[i][f] = act(agg[i][f] + b[f]);  relu optional
__global__ void bias_act_kernel(const float* __restrict__ agg,
                                const float* __restrict__ b,
                                float* __restrict__ g,
                                int N, int H, int relu)
{
    int t = blockIdx.x * blockDim.x + threadIdx.x;
    if (t >= N * H) return;
    const int f = t % H;
    float v = agg[t] + b[f];
    if (relu) v = fmaxf(v, 0.0f);
    g[t] = v;
}

// -------------------------------------------------------------------------
// Injection flags + log-softmax
// -------------------------------------------------------------------------
__global__ void flags_zero_kernel(int* __restrict__ flags, int N)
{
    int i = blockIdx.x * blockDim.x + threadIdx.x;
    if (i < N) flags[i] = 0;
}

__global__ void flags_set_kernel(const int* __restrict__ idx,
                                 int* __restrict__ flags, int M)
{
    int i = blockIdx.x * blockDim.x + threadIdx.x;
    if (i < M) flags[idx[i]] = 1;              // idempotent (dedup = .set semantics)
}

__global__ void logsoftmax_kernel(const float* __restrict__ logits,
                                  const int* __restrict__ preds,
                                  const int* __restrict__ flags,
                                  float* __restrict__ out,
                                  int N, int C, int LP /* padded stride */)
{
    int n = blockIdx.x * blockDim.x + threadIdx.x;
    if (n >= N) return;
    const float* __restrict__ row = logits + (size_t)n * LP;
    const int p = preds[n];
    const float inj = flags[n] ? 1.0f : 0.0f;  // LAM = 1.0

    float mx = -3.0e38f;
    for (int c = 0; c < C; ++c) {
        float v = row[c] + ((c == p) ? inj : 0.0f);
        mx = fmaxf(mx, v);
    }
    float s = 0.0f;
    for (int c = 0; c < C; ++c) {
        float v = row[c] + ((c == p) ? inj : 0.0f);
        s += __expf(v - mx);
    }
    const float lse = mx + __logf(s);
    float* __restrict__ orow = out + (size_t)n * C;
    for (int c = 0; c < C; ++c) {
        float v = row[c] + ((c == p) ? inj : 0.0f);
        orow[c] = v - lse;
    }
}

// -------------------------------------------------------------------------
// Host-side launch
// -------------------------------------------------------------------------
extern "C" void kernel_launch(void* const* d_in, const int* in_sizes, int n_in,
                              void* d_out, int out_size, void* d_ws, size_t ws_size,
                              hipStream_t stream)
{
    const float* x    = (const float*)d_in[0];
    const float* W1   = (const float*)d_in[1];
    const float* b1   = (const float*)d_in[2];
    const float* W2   = (const float*)d_in[3];
    const float* b2   = (const float*)d_in[4];
    const float* Wm1  = (const float*)d_in[5];
    const float* bm1  = (const float*)d_in[6];
    const float* Wm2  = (const float*)d_in[7];
    const float* bm2  = (const float*)d_in[8];
    const int*   ei    = (const int*)d_in[9];
    const int*   preds = (const int*)d_in[10];
    const int*   inj   = (const int*)d_in[11];

    const int N  = in_sizes[10];               // 100000
    const int E  = in_sizes[9] / 2;            // 1.6M
    const int NI = in_sizes[11];               // 10000
    constexpr int IN = 128;
    constexpr int H  = 64;
    constexpr int C  = 40;
    constexpr int LP = 48;                     // padded logits stride (3 col tiles)
    const int chunks = H / 4;                  // 16

    const int* src = ei;
    const int* dst = ei + E;

    // Workspace carve-out (256B aligned)
    char* cur = (char*)d_ws;
    auto carve = [&](size_t bytes) {
        void* p = (void*)cur;
        cur += (bytes + 255) & ~(size_t)255;
        return p;
    };
    float* h      = (float*)carve((size_t)N * H * 4);   // GEMM outputs / MLP hidden
    float* agg    = (float*)carve((size_t)N * H * 4);   // aggregation accumulator
    float* g      = (float*)carve((size_t)N * H * 4);   // post-bias activations
    float* logits = (float*)carve((size_t)N * LP * 4);  // padded logits
    float* deg    = (float*)carve((size_t)N * 4);
    float* dinv   = (float*)carve((size_t)N * 4);
    int*   flags  = (int*)carve((size_t)N * 4);
    (void)ws_size;

    const int T = 256;
    const int gN    = (N + T - 1) / T;
    const int gE    = (E + T - 1) / T;
    const int gNH   = (N * H + T - 1) / T;
    const int gNC4  = (N * chunks + T - 1) / T;
    const int gEC4  = (E * chunks + T - 1) / T;
    const int gNI   = (NI + T - 1) / T;
    const int gGemm = (N + 31) / 32;

    // Degree + symmetric normalization
    deg_init_kernel<<<gN, T, 0, stream>>>(deg, N);
    deg_accum_kernel<<<gE, T, 0, stream>>>(dst, deg, E);
    dinv_kernel<<<gN, T, 0, stream>>>(deg, dinv, N);

    // --- Layer 1: h = x @ W1 ; agg = D^-1/2 A D^-1/2 h ; g = relu(agg + b1)
    gemm_wmma_bf16<IN, H, H, 0><<<gGemm, T, 0, stream>>>(x, W1, nullptr, h, N);
    selfloop_init_kernel<<<gNC4, T, 0, stream>>>(h, dinv, agg, N, chunks);
    edge_scatter_kernel<<<gEC4, T, 0, stream>>>(src, dst, dinv, h, agg, E, chunks);
    bias_act_kernel<<<gNH, T, 0, stream>>>(agg, b1, g, N, H, 1);

    // --- Layer 2: h = g @ W2 ; agg = norm-aggregate ; g = agg + b2
    gemm_wmma_bf16<H, H, H, 0><<<gGemm, T, 0, stream>>>(g, W2, nullptr, h, N);
    selfloop_init_kernel<<<gNC4, T, 0, stream>>>(h, dinv, agg, N, chunks);
    edge_scatter_kernel<<<gEC4, T, 0, stream>>>(src, dst, dinv, h, agg, E, chunks);
    bias_act_kernel<<<gNH, T, 0, stream>>>(agg, b2, g, N, H, 0);

    // --- MLP head: h = relu(g @ Wm1 + bm1) ; logits = h @ Wm2 + bm2
    gemm_wmma_bf16<H, H, H, 2><<<gGemm, T, 0, stream>>>(g, Wm1, bm1, h, N);
    gemm_wmma_bf16<H, LP, C, 1><<<gGemm, T, 0, stream>>>(h, Wm2, bm2, logits, N);

    // --- Injection (assignment semantics -> dedup via flags) + log-softmax
    flags_zero_kernel<<<gN, T, 0, stream>>>(flags, N);
    flags_set_kernel<<<gNI, T, 0, stream>>>(inj, flags, NI);
    logsoftmax_kernel<<<gN, T, 0, stream>>>(logits, preds, flags, (float*)d_out,
                                            N, C, LP);
}
